// RTERModel_78323023610576
// MI455X (gfx1250) — matrix operand order repaired and checked by reading, not compiled, not run on hardware
//
#include <hip/hip_runtime.h>
#include <hip/hip_bf16.h>
#include <math.h>

typedef __bf16 bf16_t;
typedef __attribute__((ext_vector_type(16))) __bf16 v16bf;
typedef __attribute__((ext_vector_type(8)))  __bf16 v8bf;
typedef __attribute__((ext_vector_type(8)))  float  v8f;

// ---- static model config ----
#define NN_   128
#define LL    64
#define EE    300
#define HH    300
#define KEFF_ 40
#define HOPS_ 3
#define NCLS_ 6
#define G3    900     // 3*H
#define EPAD  320     // K padding (mult of 32)
#define HPAD  320
#define H2PAD 608
#define NQ    127     // N-1
// padded output (N) dims, mult of 32 so NT=2 tiles divide evenly
#define G3P   928     // 900 -> 928
#define HOP   320     // 300 -> 320
#define H2OP  608     // 600 -> 608
#define MPQ   5088    // 40*127=5080 -> 5088 (mult of 16)

__device__ __forceinline__ float sigmoidf_(float x) { return 1.0f / (1.0f + __expf(-x)); }

// =================== WMMA GEMM: C[Mp,Npad] = A[Mp,KP] * Bswz + biasPad ===================
// A: row-major bf16, lda mult of 16 elems (16B-aligned rows), K zero-padded.
// Bswz: fragment-swizzled bf16: [(ntile*K32+kb)*32 + lane][16] contiguous per lane.
// C: f32 row-major with padded ldc; bias: zero-padded to Npad. Fully unconditional.
template <int KP, int NT>
__global__ __launch_bounds__(128)
void k_gemm(const bf16_t* __restrict__ A, int lda,
            const bf16_t* __restrict__ Bs,
            float* __restrict__ C, int ldc,
            const float* __restrict__ bias, int Mp)
{
    constexpr int K32 = KP / 32;
    const int lane = threadIdx.x & 31;
    const int wave = threadIdx.x >> 5;
    const int mt   = blockIdx.y * 4 + wave;
    if (mt * 16 >= Mp) return;                 // uniform per wave
    const int nt0  = blockIdx.x * NT;
    const int hh   = lane >> 4;
    const int l16  = lane & 15;
    const int arow = mt * 16 + l16;

    v8f acc[NT];
#pragma unroll
    for (int n = 0; n < NT; ++n) acc[n] = (v8f){};

    const bf16_t* Ap = A + (size_t)arow * lda + 8 * hh;
    const bf16_t* Bp = Bs + ((size_t)nt0 * K32 * 32 + lane) * 16;

#pragma unroll
    for (int kb = 0; kb < K32; ++kb) {
        v8bf alo = *(const v8bf*)(Ap + kb * 32);
        v8bf ahi = *(const v8bf*)(Ap + kb * 32 + 16);
        v16bf a = __builtin_shufflevector(alo, ahi, 0, 1, 2, 3, 4, 5, 6, 7,
                                          8, 9, 10, 11, 12, 13, 14, 15);
#pragma unroll
        for (int n = 0; n < NT; ++n) {
            v16bf b = *(const v16bf*)(Bp + (size_t)(n * K32 + kb) * 512);
            acc[n] = __builtin_amdgcn_wmma_f32_16x16x32_bf16(false, a, false, b,
                                                             (short)0, acc[n], false, false);
        }
    }
#pragma unroll
    for (int n = 0; n < NT; ++n) {
        const int col = (nt0 + n) * 16 + l16;
        const float bb = bias[col];
#pragma unroll
        for (int r = 0; r < 8; ++r) {
            const int row = mt * 16 + r + 8 * hh;
            C[(size_t)row * ldc + col] = acc[n][r] + bb;
        }
    }
}

// =================== weight prep: f32 [Nout, K] -> fragment-swizzled bf16 ===================
__global__ void k_swizzleB(const float* __restrict__ W, bf16_t* __restrict__ out,
                           int Nout, int K, int K32, int Ntiles)
{
    int i = blockIdx.x * blockDim.x + threadIdx.x;
    int total = Ntiles * K32 * 512;
    if (i >= total) return;
    int e = i & 15, lane = (i >> 4) & 31, t = i >> 9;
    int kb = t % K32, nt = t / K32;
    int n = nt * 16 + (lane & 15);
    int k = kb * 32 + (lane >> 4) * 8 + (e & 7) + (e >> 3) * 16;
    float v = (n < Nout && k < K) ? W[(size_t)n * K + k] : 0.0f;
    out[i] = (bf16_t)v;
}

// two stacked matrices -> swizzled B of [W1; W2] columns (Nout = 2n)
__global__ void k_swizzleB2(const float* __restrict__ W1, const float* __restrict__ W2,
                            bf16_t* __restrict__ out, int n1, int K, int K32, int Ntiles)
{
    int i = blockIdx.x * blockDim.x + threadIdx.x;
    int total = Ntiles * K32 * 512;
    if (i >= total) return;
    int e = i & 15, lane = (i >> 4) & 31, t = i >> 9;
    int kb = t % K32, nt = t / K32;
    int n = nt * 16 + (lane & 15);
    int k = kb * 32 + (lane >> 4) * 8 + (e & 7) + (e >> 3) * 16;
    float v = 0.0f;
    if (k < K && n < 2 * n1)
        v = (n < n1) ? W1[(size_t)n * K + k] : W2[(size_t)(n - n1) * K + k];
    out[i] = (bf16_t)v;
}

__global__ void k_pad_bias(const float* __restrict__ src, float* __restrict__ dst, int n, int np)
{ int i = blockIdx.x * blockDim.x + threadIdx.x; if (i < np) dst[i] = (i < n) ? src[i] : 0.0f; }

__global__ void k_pad_bias2(const float* __restrict__ b1, const float* __restrict__ b2,
                            float* __restrict__ dst, int n, int np)
{
    int i = blockIdx.x * blockDim.x + threadIdx.x;
    if (i >= np) return;
    dst[i] = (i < n) ? b1[i] : (i < 2 * n ? b2[i - n] : 0.0f);
}

__global__ void k_zero_f32(float* p, int n)
{ int i = blockIdx.x * blockDim.x + threadIdx.x; if (i < n) p[i] = 0.0f; }

__global__ void k_zero_bf16(bf16_t* p, int n)
{ int i = blockIdx.x * blockDim.x + threadIdx.x; if (i < n) p[i] = (bf16_t)0.0f; }

// =================== embedding gather (fwd + length-reversed), f32 -> bf16, K-padded ===================
__global__ void k_embed(const int* __restrict__ ids, const int* __restrict__ lens,
                        const float* __restrict__ emb,
                        bf16_t* __restrict__ xf, bf16_t* __restrict__ xr)
{
    int i = blockIdx.x * blockDim.x + threadIdx.x;
    int total = LL * NN_ * EPAD;
    if (i >= total) return;
    int t = i / (NN_ * EPAD);
    int r = i % (NN_ * EPAD);
    int b = r / EPAD;
    int e = r % EPAD;
    float vf = 0.0f, vr = 0.0f;
    if (e < EE) {
        int idf = ids[b * LL + t];
        vf = emb[(size_t)idf * EE + e];
        int rt = lens[b] - 1 - t;
        rt = rt < 0 ? 0 : (rt > LL - 1 ? LL - 1 : rt);
        int idr = ids[b * LL + rt];
        vr = emb[(size_t)idr * EE + e];
    }
    xf[i] = (bf16_t)vf;
    xr[i] = (bf16_t)vr;
}

// =================== GRU gate update (r,z,n order), masked ===================
__global__ void k_gru_gate(const float* __restrict__ gi_t,  // row stride G3P (incl. bih)
                           const float* __restrict__ gh,    // row stride G3P (incl. bhh)
                           float* __restrict__ h,           // [B,300]
                           bf16_t* __restrict__ hb,         // [128,320] (pads pre-zeroed)
                           float* __restrict__ out_t,       // [B,300]
                           const int* __restrict__ lens, int t, int B)
{
    int i = blockIdx.x * blockDim.x + threadIdx.x;
    if (i >= B * HH) return;
    int b = i / HH, j = i % HH;
    float ir = gi_t[(size_t)b * G3P + j];
    float iz = gi_t[(size_t)b * G3P + HH + j];
    float in = gi_t[(size_t)b * G3P + 2 * HH + j];
    float hr = gh[(size_t)b * G3P + j];
    float hz = gh[(size_t)b * G3P + HH + j];
    float hn = gh[(size_t)b * G3P + 2 * HH + j];
    float rr = sigmoidf_(ir + hr);
    float zz = sigmoidf_(iz + hz);
    float nn = tanhf(in + rr * hn);
    float hprev = h[i];
    float hnew = (1.0f - zz) * nn + zz * hprev;
    bool valid = lens ? (t < lens[b]) : true;
    float hv = valid ? hnew : hprev;
    h[i] = hv;
    hb[(size_t)b * HPAD + j] = (bf16_t)hv;
    out_t[i] = valid ? hnew : 0.0f;
}

// =================== temporal max-pool -> bf16 [128, 608] ===================
__global__ void k_maxpool(const float* __restrict__ outf, const float* __restrict__ outr,
                          bf16_t* __restrict__ mp)
{
    int i = blockIdx.x * blockDim.x + threadIdx.x;
    if (i >= NN_ * H2PAD) return;
    int b = i / H2PAD, j = i % H2PAD;
    float m = 0.0f;
    if (j < 2 * HH) {
        const float* src = (j < HH) ? outf : outr;
        int jj = (j < HH) ? j : j - HH;
        m = -3.4e38f;
        for (int t = 0; t < LL; ++t) m = fmaxf(m, src[(size_t)t * NN_ * HH + b * HH + jj]);
    }
    mp[i] = (bf16_t)m;
}

__global__ void k_tanh_inplace(float* __restrict__ x, int n)
{ int i = blockIdx.x * blockDim.x + threadIdx.x; if (i < n) x[i] = tanhf(x[i]); }

// =================== causal context windows (u row stride HOP) ===================
__global__ void k_build_ctx(const float* __restrict__ u,
                            bf16_t* __restrict__ ctxb, float* __restrict__ ctxf)
{
    int i = blockIdx.x * blockDim.x + threadIdx.x;
    int total = MPQ * HPAD;
    if (i >= total) return;
    int row = i / HPAD, e = i % HPAD;
    float v = 0.0f;
    if (row < KEFF_ * NQ) {
        int k = row / NQ, q = row % NQ;
        int src = (q + 1) - KEFF_ + k;
        if (src >= 0 && e < HH) v = u[(size_t)src * HOP + e];
        if (e < HH) ctxf[(size_t)row * HH + e] = v;
    }
    ctxb[i] = (bf16_t)v;
}

// =================== inp = ctx + mem_fwd + un-reversed mem_bwd ===================
__global__ void k_mem_bank(const float* __restrict__ ctxf, const float* __restrict__ memf,
                           const float* __restrict__ memrev,
                           float* __restrict__ inpf, bf16_t* __restrict__ inpb)
{
    int i = blockIdx.x * blockDim.x + threadIdx.x;
    int total = MPQ * HPAD;
    if (i >= total) return;
    int row = i / HPAD, e = i % HPAD;
    float v = 0.0f;
    if (row < KEFF_ * NQ && e < HH) {
        int k = row / NQ, q = row % NQ;
        size_t idx = (size_t)row * HH + e;
        size_t ridx = (size_t)((KEFF_ - 1 - k) * NQ + q) * HH + e;
        v = ctxf[idx] + memf[idx] + memrev[ridx];
        inpf[idx] = v;
    }
    inpb[i] = (bf16_t)v;
}

// =================== attention scores + masked softmax -> gates[k, q] ===================
__global__ void k_attn_softmax(const float* __restrict__ query, const float* __restrict__ inpf,
                               float* __restrict__ gates)
{
    __shared__ float s[KEFF_];
    int q = blockIdx.x;
    int k = threadIdx.x;
    if (k < KEFF_) {
        float d = 0.0f;
        const float* qp = query + (size_t)q * HH;
        const float* mp = inpf + (size_t)(k * NQ + q) * HH;
        for (int j = 0; j < HH; ++j) d += qp[j] * mp[j];
        s[k] = (k < (KEFF_ - 1 - q)) ? -1e30f : d;
    }
    __syncthreads();
    if (threadIdx.x == 0) {
        float mx = -3.4e38f;
        for (int kk = 0; kk < KEFF_; ++kk) mx = fmaxf(mx, s[kk]);
        float sum = 0.0f;
        for (int kk = 0; kk < KEFF_; ++kk) { s[kk] = __expf(s[kk] - mx); sum += s[kk]; }
        float inv = 1.0f / sum;
        for (int kk = 0; kk < KEFF_; ++kk) gates[kk * NQ + q] = s[kk] * inv;
    }
}

// =================== AttnGRU cell step ===================
__global__ void k_att_gate(const float* __restrict__ xr_k,   // row stride HOP
                           const float* __restrict__ xw_k,   // row stride HOP
                           const float* __restrict__ uh,     // row stride H2OP ([Ur|Uw]h + biases)
                           const float* __restrict__ gates_k,
                           float* __restrict__ h, bf16_t* __restrict__ hb)
{
    int i = blockIdx.x * blockDim.x + threadIdx.x;
    if (i >= NQ * HH) return;
    int q = i / HH, j = i % HH;
    float rr = sigmoidf_(xr_k[(size_t)q * HOP + j] + uh[(size_t)q * H2OP + j]);
    float ht = tanhf(xw_k[(size_t)q * HOP + j] + rr * uh[(size_t)q * H2OP + HH + j]);
    float g = gates_k[q];
    float hn = g * ht + (1.0f - g) * h[i];
    h[i] = hn;
    hb[(size_t)q * HPAD + j] = (bf16_t)hn;
}

__global__ void k_init_query(const float* __restrict__ u, float* __restrict__ query)
{
    int i = blockIdx.x * blockDim.x + threadIdx.x;
    if (i >= NQ * HH) return;
    int q = i / HH, j = i % HH;
    query[i] = u[(size_t)(q + 1) * HOP + j];
}

__global__ void k_query_update(float* __restrict__ query, const float* __restrict__ hf,
                               const float* __restrict__ hb)
{
    int i = blockIdx.x * blockDim.x + threadIdx.x;
    if (i < NQ * HH) query[i] += hf[i] + hb[i];
}

// =================== classifier ===================
__global__ void k_cls(const float* __restrict__ u, const float* __restrict__ query,
                      const float* __restrict__ w, const float* __restrict__ bias,
                      float* __restrict__ out)
{
    int n = blockIdx.x;
    int c = threadIdx.x;
    if (c >= NCLS_) return;
    const float* row = (n == 0) ? u : (query + (size_t)(n - 1) * HH);
    float acc = bias[c];
    for (int j = 0; j < HH; ++j) acc += row[j] * w[(size_t)c * HH + j];
    out[(size_t)n * NCLS_ + c] = acc;
}

// =====================================================================================

extern "C" void kernel_launch(void* const* d_in, const int* in_sizes, int n_in,
                              void* d_out, int out_size, void* d_ws, size_t ws_size,
                              hipStream_t stream)
{
    const int*   ids      = (const int*)d_in[0];
    const int*   lens     = (const int*)d_in[1];
    const float* emb      = (const float*)d_in[2];
    const float* utt_Wih  = (const float*)d_in[3];
    const float* utt_Whh  = (const float*)d_in[4];
    const float* utt_bih  = (const float*)d_in[5];
    const float* utt_bhh  = (const float*)d_in[6];
    const float* lin_w    = (const float*)d_in[7];
    const float* lin_b    = (const float*)d_in[8];
    const float* ctx_Wih  = (const float*)d_in[9];
    const float* ctx_Whh  = (const float*)d_in[10];
    const float* ctx_bih  = (const float*)d_in[11];
    const float* ctx_bhh  = (const float*)d_in[12];
    const float* att_Wr_w = (const float*)d_in[13];
    const float* att_Wr_b = (const float*)d_in[14];
    const float* att_Ur_w = (const float*)d_in[15];
    const float* att_Ur_b = (const float*)d_in[16];
    const float* att_W_w  = (const float*)d_in[17];
    const float* att_W_b  = (const float*)d_in[18];
    const float* att_U_w  = (const float*)d_in[19];
    const float* att_U_b  = (const float*)d_in[20];
    const float* cls_w    = (const float*)d_in[21];
    const float* cls_b    = (const float*)d_in[22];
    float* out = (float*)d_out;

    // ---- workspace bump allocation (fixed, deterministic) ----
    char* base = (char*)d_ws;
    size_t off = 0;
    auto alloc = [&](size_t bytes) -> void* {
        void* p = base + off;
        off = (off + bytes + 255) & ~(size_t)255;
        return p;
    };
    // swizzled bf16 weights (size = Kpad * Npad each)
    bf16_t* uttWihS = (bf16_t*)alloc((size_t)2 * EPAD * G3P * 2);
    bf16_t* uttWhhS = (bf16_t*)alloc((size_t)2 * HPAD * G3P * 2);
    bf16_t* linWS   = (bf16_t*)alloc((size_t)H2PAD * HOP * 2);
    bf16_t* ctxWihS = (bf16_t*)alloc((size_t)2 * HPAD * G3P * 2);
    bf16_t* ctxWhhS = (bf16_t*)alloc((size_t)2 * HPAD * G3P * 2);
    bf16_t* WrS     = (bf16_t*)alloc((size_t)6 * HPAD * HOP * 2);
    bf16_t* WwS     = (bf16_t*)alloc((size_t)6 * HPAD * HOP * 2);
    bf16_t* US      = (bf16_t*)alloc((size_t)6 * HPAD * H2OP * 2);
    // padded f32 biases
    float* bihP  = (float*)alloc((size_t)2 * G3P * 4);
    float* bhhP  = (float*)alloc((size_t)2 * G3P * 4);
    float* cbihP = (float*)alloc((size_t)2 * G3P * 4);
    float* cbhhP = (float*)alloc((size_t)2 * G3P * 4);
    float* linbP = (float*)alloc((size_t)HOP * 4);
    float* wrbP  = (float*)alloc((size_t)6 * HOP * 4);
    float* wwbP  = (float*)alloc((size_t)6 * HOP * 4);
    float* ubP   = (float*)alloc((size_t)6 * H2OP * 4);
    // activations
    bf16_t* xf    = (bf16_t*)alloc((size_t)LL * NN_ * EPAD * 2);
    bf16_t* xr    = (bf16_t*)alloc((size_t)LL * NN_ * EPAD * 2);
    float*  gi    = (float*)alloc((size_t)LL * NN_ * G3P * 4);   // reused for ctx gi (5088 rows)
    float*  gh    = (float*)alloc((size_t)NN_ * G3P * 4);        // also reused as uh [128,608]
    float*  hA    = (float*)alloc((size_t)NN_ * HH * 4);
    bf16_t* hAb   = (bf16_t*)alloc((size_t)NN_ * HPAD * 2);
    float*  hB    = (float*)alloc((size_t)NN_ * HH * 4);
    bf16_t* hBb   = (bf16_t*)alloc((size_t)NN_ * HPAD * 2);
    float*  outF  = (float*)alloc((size_t)LL * NN_ * HH * 4);
    float*  outR  = (float*)alloc((size_t)LL * NN_ * HH * 4);
    bf16_t* mp    = (bf16_t*)alloc((size_t)NN_ * H2PAD * 2);
    float*  u     = (float*)alloc((size_t)NN_ * HOP * 4);
    bf16_t* ctxb  = (bf16_t*)alloc((size_t)MPQ * HPAD * 2);
    float*  ctxf  = (float*)alloc((size_t)KEFF_ * NQ * HH * 4);
    float*  memF  = (float*)alloc((size_t)KEFF_ * NQ * HH * 4);
    float*  memR  = (float*)alloc((size_t)KEFF_ * NQ * HH * 4);
    float*  inpf  = (float*)alloc((size_t)KEFF_ * NQ * HH * 4);
    bf16_t* inpb  = (bf16_t*)alloc((size_t)MPQ * HPAD * 2);
    float*  gates = (float*)alloc((size_t)KEFF_ * NQ * 4);
    float*  xrPre = (float*)alloc((size_t)MPQ * HOP * 4);
    float*  xwPre = (float*)alloc((size_t)MPQ * HOP * 4);
    float*  query = (float*)alloc((size_t)NQ * HH * 4);

    auto cdiv = [](int a, int b) { return (a + b - 1) / b; };
    // GEMM dispatchers (KP compile-time, NT=2)
    auto gemm320 = [&](const bf16_t* A, int lda, const bf16_t* Bs, float* C, int ldc,
                       const float* bias, int Mp, int Npad) {
        dim3 grid(Npad / 32, cdiv(Mp, 64));
        k_gemm<320, 2><<<grid, 128, 0, stream>>>(A, lda, Bs, C, ldc, bias, Mp);
    };
    auto gemm608 = [&](const bf16_t* A, int lda, const bf16_t* Bs, float* C, int ldc,
                       const float* bias, int Mp, int Npad) {
        dim3 grid(Npad / 32, cdiv(Mp, 64));
        k_gemm<608, 2><<<grid, 128, 0, stream>>>(A, lda, Bs, C, ldc, bias, Mp);
    };
    auto swz = [&](const float* W, bf16_t* o, int Nout, int K, int Kpad, int Npad) {
        int tot = (Npad / 16) * (Kpad / 32) * 512;
        k_swizzleB<<<cdiv(tot, 256), 256, 0, stream>>>(W, o, Nout, K, Kpad / 32, Npad / 16);
    };
    auto padb = [&](const float* s, float* d, int n, int np) {
        k_pad_bias<<<cdiv(np, 256), 256, 0, stream>>>(s, d, n, np);
    };
    auto zeroF = [&](float* p, int n) { k_zero_f32<<<cdiv(n, 256), 256, 0, stream>>>(p, n); };
    auto zeroB = [&](bf16_t* p, int n) { k_zero_bf16<<<cdiv(n, 256), 256, 0, stream>>>(p, n); };

    // ---- weight + bias preparation ----
    for (int d = 0; d < 2; ++d) {
        swz(utt_Wih + (size_t)d * G3 * EE, uttWihS + (size_t)d * EPAD * G3P, G3, EE, EPAD, G3P);
        swz(utt_Whh + (size_t)d * G3 * HH, uttWhhS + (size_t)d * HPAD * G3P, G3, HH, HPAD, G3P);
        swz(ctx_Wih + (size_t)d * G3 * HH, ctxWihS + (size_t)d * HPAD * G3P, G3, HH, HPAD, G3P);
        swz(ctx_Whh + (size_t)d * G3 * HH, ctxWhhS + (size_t)d * HPAD * G3P, G3, HH, HPAD, G3P);
        padb(utt_bih + (size_t)d * G3, bihP + (size_t)d * G3P, G3, G3P);
        padb(utt_bhh + (size_t)d * G3, bhhP + (size_t)d * G3P, G3, G3P);
        padb(ctx_bih + (size_t)d * G3, cbihP + (size_t)d * G3P, G3, G3P);
        padb(ctx_bhh + (size_t)d * G3, cbhhP + (size_t)d * G3P, G3, G3P);
    }
    swz(lin_w, linWS, HH, 2 * HH, H2PAD, HOP);
    padb(lin_b, linbP, HH, HOP);
    for (int hd = 0; hd < 6; ++hd) {
        swz(att_Wr_w + (size_t)hd * HH * HH, WrS + (size_t)hd * HPAD * HOP, HH, HH, HPAD, HOP);
        swz(att_W_w + (size_t)hd * HH * HH, WwS + (size_t)hd * HPAD * HOP, HH, HH, HPAD, HOP);
        int tot = (H2OP / 16) * (HPAD / 32) * 512;
        k_swizzleB2<<<cdiv(tot, 256), 256, 0, stream>>>(
            att_Ur_w + (size_t)hd * HH * HH, att_U_w + (size_t)hd * HH * HH,
            US + (size_t)hd * HPAD * H2OP, HH, HH, HPAD / 32, H2OP / 16);
        padb(att_Wr_b + (size_t)hd * HH, wrbP + (size_t)hd * HOP, HH, HOP);
        padb(att_W_b + (size_t)hd * HH, wwbP + (size_t)hd * HOP, HH, HOP);
        k_pad_bias2<<<cdiv(H2OP, 256), 256, 0, stream>>>(
            att_Ur_b + (size_t)hd * HH, att_U_b + (size_t)hd * HH, ubP + (size_t)hd * H2OP, HH, H2OP);
    }

    // ---- embedding gather (fwd + reversed) ----
    { int tot = LL * NN_ * EPAD; k_embed<<<cdiv(tot, 256), 256, 0, stream>>>(ids, lens, emb, xf, xr); }

    // ---- utterance BiGRU ----
    for (int d = 0; d < 2; ++d) {
        const bf16_t* x = d ? xr : xf;
        gemm320(x, EPAD, uttWihS + (size_t)d * EPAD * G3P, gi, G3P,
                bihP + (size_t)d * G3P, LL * NN_, G3P);
        zeroF(hA, NN_ * HH); zeroB(hAb, NN_ * HPAD);
        float* o = d ? outR : outF;
        for (int t = 0; t < LL; ++t) {
            gemm320(hAb, HPAD, uttWhhS + (size_t)d * HPAD * G3P, gh, G3P,
                    bhhP + (size_t)d * G3P, NN_, G3P);
            k_gru_gate<<<cdiv(NN_ * HH, 256), 256, 0, stream>>>(
                gi + (size_t)t * NN_ * G3P, gh, hA, hAb, o + (size_t)t * NN_ * HH, lens, t, NN_);
        }
    }

    // ---- max-pool + linear + tanh ----
    k_maxpool<<<cdiv(NN_ * H2PAD, 256), 256, 0, stream>>>(outF, outR, mp);
    gemm608(mp, H2PAD, linWS, u, HOP, linbP, NN_, HOP);
    k_tanh_inplace<<<cdiv(NN_ * HOP, 256), 256, 0, stream>>>(u, NN_ * HOP);

    // ---- causal context windows + memory BiGRU ----
    { int tot = MPQ * HPAD; k_build_ctx<<<cdiv(tot, 256), 256, 0, stream>>>(u, ctxb, ctxf); }
    for (int d = 0; d < 2; ++d) {
        gemm320(ctxb, HPAD, ctxWihS + (size_t)d * HPAD * G3P, gi, G3P,
                cbihP + (size_t)d * G3P, MPQ, G3P);
        zeroF(hA, NN_ * HH); zeroB(hAb, NN_ * HPAD);
        float* o = d ? memR : memF;
        for (int t = 0; t < KEFF_; ++t) {
            int ti = d ? (KEFF_ - 1 - t) : t;   // backward direction reads reversed gi rows
            gemm320(hAb, HPAD, ctxWhhS + (size_t)d * HPAD * G3P, gh, G3P,
                    cbhhP + (size_t)d * G3P, NN_, G3P);
            k_gru_gate<<<cdiv(NQ * HH, 256), 256, 0, stream>>>(
                gi + (size_t)ti * NQ * G3P, gh, hA, hAb, o + (size_t)t * NQ * HH, nullptr, t, NQ);
        }
    }
    { int tot = MPQ * HPAD; k_mem_bank<<<cdiv(tot, 256), 256, 0, stream>>>(ctxf, memF, memR, inpf, inpb); }

    // ---- attention hops ----
    k_init_query<<<cdiv(NQ * HH, 256), 256, 0, stream>>>(u, query);
    for (int hop = 0; hop < HOPS_; ++hop) {
        k_attn_softmax<<<NQ, 64, 0, stream>>>(query, inpf, gates);
        for (int d = 0; d < 2; ++d) {
            int hd = hop * 2 + d;
            gemm320(inpb, HPAD, WrS + (size_t)hd * HPAD * HOP, xrPre, HOP,
                    wrbP + (size_t)hd * HOP, MPQ, HOP);
            gemm320(inpb, HPAD, WwS + (size_t)hd * HPAD * HOP, xwPre, HOP,
                    wwbP + (size_t)hd * HOP, MPQ, HOP);
            float*  h  = d ? hB : hA;
            bf16_t* hb = d ? hBb : hAb;
            zeroF(h, NN_ * HH); zeroB(hb, NN_ * HPAD);
            for (int s = 0; s < KEFF_; ++s) {
                int k = d ? (KEFF_ - 1 - s) : s;
                // fused uh = [h@Ur^T+bur | h@Uw^T+bu] -> gh reused as [128,608]
                gemm320(hb, HPAD, US + (size_t)hd * HPAD * H2OP, gh, H2OP,
                        ubP + (size_t)hd * H2OP, NN_, H2OP);
                k_att_gate<<<cdiv(NQ * HH, 256), 256, 0, stream>>>(
                    xrPre + (size_t)k * NQ * HOP, xwPre + (size_t)k * NQ * HOP, gh,
                    gates + (size_t)k * NQ, h, hb);
            }
        }
        k_query_update<<<cdiv(NQ * HH, 256), 256, 0, stream>>>(query, hA, hB);
    }

    // ---- classifier ----
    k_cls<<<NN_, 32, 0, stream>>>(u, query, cls_w, cls_b, out);

    (void)in_sizes; (void)n_in; (void)out_size; (void)ws_size;
}
